// ClassicalLSTMCell_65481071399014
// MI455X (gfx1250) — matrix operand order repaired and checked by reading, not compiled, not run on hardware
//
#include <hip/hip_runtime.h>
#include <stdint.h>

// ---------------------------------------------------------------------------
// Persistent cooperative LSTM for MI455X (gfx1250).
//   T=1024, B=128, D_IN=256, D_H=256, K = 512.
//   16 workgroups (1 per WGP), 256 threads (8 wave32) each.
//   WG w owns hidden columns [16w,16w+16) for all four gates.
//   Wave m owns batch rows [16m,16m+16).  c-state lives in VGPRs for all T.
//   Weights (bf16, transposed [g][n][k]) live in LDS for all T.
//   h is broadcast between WGPs through a double-buffered global bf16 buffer
//   with one device-scope atomic barrier per timestep.
// ---------------------------------------------------------------------------

#define T_STEPS 1024
#define BATCH   128
#define DIN     256
#define DH      256
#define KDIM    (DIN + DH)        // 512
#define NWG     16
#define THREADS 256               // 8 wave32
#define APAD    8                 // pad halves per LDS row (bank spread)
#define AROW    (KDIM + APAD)     // 520 halves = 1040 B per row
#define WROW    (KDIM + APAD)

typedef __attribute__((ext_vector_type(8)))  float  v8f;
typedef __attribute__((ext_vector_type(8)))  __bf16 v8bf;
typedef __attribute__((ext_vector_type(16))) __bf16 v16bf;
typedef __attribute__((ext_vector_type(4)))  int    v4i;

__device__ __forceinline__ unsigned short f2bf(float f) {
  unsigned u = __float_as_uint(f);
  u += 0x7FFFu + ((u >> 16) & 1u);          // round-to-nearest-even
  return (unsigned short)(u >> 16);
}
__device__ __forceinline__ float sigm(float x)   { return 1.0f / (1.0f + __expf(-x)); }
__device__ __forceinline__ float tanh_f(float x) { return 2.0f / (1.0f + __expf(-2.0f * x)) - 1.0f; }

__device__ __forceinline__ v8f splat8(float x) {
  v8f v;
#pragma unroll
  for (int i = 0; i < 8; ++i) v[i] = x;
  return v;
}

// Load one 16-half WMMA operand fragment from LDS as two b128 chunks.
__device__ __forceinline__ v16bf ld_frag(const unsigned short* p0,
                                         const unsigned short* p1) {
  v8bf lo = *(const v8bf*)p0;               // ds_load_b128
  v8bf hi = *(const v8bf*)p1;               // ds_load_b128
  return __builtin_shufflevector(lo, hi, 0, 1, 2, 3, 4, 5, 6, 7,
                                         8, 9, 10, 11, 12, 13, 14, 15);
}

// ---------------------------------------------------------------------------
// Kernel 1: convert & transpose weights fp32 [512,256] -> bf16 [4][256][512],
//           zero the double-buffered h broadcast buffer and barrier state.
// ---------------------------------------------------------------------------
__global__ void lstm_prep(const float* __restrict__ Wf, const float* __restrict__ Wi,
                          const float* __restrict__ Wg, const float* __restrict__ Wo,
                          unsigned short* __restrict__ wt,
                          unsigned int* __restrict__ hb32,
                          int* __restrict__ syncv) {
  const int stride = gridDim.x * blockDim.x;
  const int idx = blockIdx.x * blockDim.x + threadIdx.x;
  for (int i = idx; i < 4 * 256 * 512; i += stride) {
    int g   = i >> 17;                      // / (256*512)
    int rem = i & 131071;
    int n   = rem >> 9;                     // / 512
    int k   = rem & 511;
    const float* W = (g == 0) ? Wf : (g == 1) ? Wi : (g == 2) ? Wg : Wo;
    wt[i] = f2bf(W[k * 256 + n]);           // wt[g][n][k] = W[k][n]
  }
  for (int i = idx; i < (2 * BATCH * DH) / 2; i += stride) hb32[i] = 0u; // h0 = 0
  if (idx < 2) syncv[idx] = 0;              // {arrive count, phase}
}

// Device-scope barrier across the NWG cooperating workgroups.
__device__ __forceinline__ void grid_barrier(int* cnt, int* phase, int want) {
  __syncthreads();
  if (threadIdx.x == 0) {
    __threadfence();                        // release h stores to device scope
    int old = __hip_atomic_fetch_add(cnt, 1, __ATOMIC_ACQ_REL,
                                     __HIP_MEMORY_SCOPE_AGENT);
    if (old == NWG - 1) {
      __hip_atomic_store(cnt, 0, __ATOMIC_RELAXED, __HIP_MEMORY_SCOPE_AGENT);
      __hip_atomic_fetch_add(phase, 1, __ATOMIC_RELEASE,
                             __HIP_MEMORY_SCOPE_AGENT);
    }
    while (__hip_atomic_load(phase, __ATOMIC_ACQUIRE,
                             __HIP_MEMORY_SCOPE_AGENT) < want) {
      __builtin_amdgcn_s_sleep(2);
    }
  }
  __syncthreads();
  __threadfence();                          // acquire for all threads
}

// ---------------------------------------------------------------------------
// Kernel 2: persistent LSTM.
// ---------------------------------------------------------------------------
__global__ __launch_bounds__(THREADS, 1)
void lstm_main(const float* __restrict__ x,
               const float* __restrict__ bF, const float* __restrict__ bI,
               const float* __restrict__ bG, const float* __restrict__ bO,
               const unsigned short* __restrict__ wt,
               unsigned short* __restrict__ hbuf,
               int* __restrict__ syncv,
               float* __restrict__ out) {
  // 195 KB static LDS — legal only because CDNA5 WGP has 320 KB.
  __shared__ __align__(16) unsigned short Ash[BATCH * AROW];   // 130 KB: [x|h] bf16
  __shared__ __align__(16) unsigned short Wsh[4 * 16 * WROW];  //  65 KB: weight slice

  const int wgid = blockIdx.x;              // 0..15 -> hidden cols [16w,16w+16)
  const int tid  = threadIdx.x;
  const int wid  = tid >> 5;                // wave id = M-tile (rows 16*wid..)
  const int lane = tid & 31;
  const int colL = lane & 15;               // N within tile
  const int col  = wgid * 16 + colL;        // global hidden column
  const int rowBase = wid * 16 + ((lane >> 4) << 3);

  // Load this WG's weight slice into LDS once (resident for all 1024 steps).
  for (int i = tid; i < 4 * 16 * KDIM; i += THREADS) {
    int g = i >> 13;                        // / (16*512)
    int rem = i & 8191;
    int n = rem >> 9;
    int k = rem & 511;
    Wsh[(g * 16 + n) * WROW + k] = wt[(g * 256 + wgid * 16 + n) * KDIM + k];
  }
  __syncthreads();

  const float biasF = bF[col], biasI = bI[col], biasG = bG[col], biasO = bO[col];

  v8f cfrag = splat8(0.0f);                 // c-state: registers for the whole run
  v8f hlast = splat8(0.0f);

  const unsigned short* aRow = &Ash[(wid * 16 + colL) * AROW];
  const int aOff = (lane >> 4) * 8;         // A: K chunks [b,b+8)+[b+16,b+24)
  const int bOff = (lane >> 4) * 8;         // B: K chunks [b,b+8)+[b+16,b+24)

  for (int t = 0; t < T_STEPS; ++t) {
    const float* xt = x + (size_t)t * (BATCH * DIN);
    const unsigned short* hcur = hbuf + (t & 1) * (BATCH * DH);
    unsigned short*       hnxt = hbuf + ((t + 1) & 1) * (BATCH * DH);

    // --- stage A = [x_t (fp32->bf16) | h_t (bf16)] into LDS, padded rows ---
    for (int i = tid; i < BATCH * (DIN / 4); i += THREADS) {
      int r = i >> 6, c4 = i & 63;
      float4 xv = ((const float4*)(xt + r * DIN))[c4];
      uint2 p;
      p.x = (unsigned)f2bf(xv.x) | ((unsigned)f2bf(xv.y) << 16);
      p.y = (unsigned)f2bf(xv.z) | ((unsigned)f2bf(xv.w) << 16);
      *(uint2*)&Ash[r * AROW + c4 * 4] = p;
    }
#if defined(__gfx1250__) && __has_builtin(__builtin_amdgcn_global_load_async_to_lds_b128)
    for (int i = tid; i < BATCH * (DH / 8); i += THREADS) {
      int r = i >> 5, c8 = i & 31;
      __builtin_amdgcn_global_load_async_to_lds_b128(
          (__attribute__((address_space(1))) v4i*)(hcur + r * DH + c8 * 8),
          (__attribute__((address_space(3))) v4i*)&Ash[r * AROW + DIN + c8 * 8],
          0, 0);
    }
    asm volatile("s_wait_asynccnt 0" ::: "memory");
#else
    for (int i = tid; i < BATCH * (DH / 8); i += THREADS) {
      int r = i >> 5, c8 = i & 31;
      *(uint4*)&Ash[r * AROW + DIN + c8 * 8] =
          *(const uint4*)(hcur + r * DH + c8 * 8);
    }
#endif
    if (t + 1 < T_STEPS)
      __builtin_prefetch(xt + BATCH * DIN + tid * 128, 0, 1); // global_prefetch_b8
    __syncthreads();

    // --- four gate GEMMs: 16 K-tiles x 4 WMMAs (independent accum chains) ---
    v8f accF = splat8(biasF), accI = splat8(biasI);
    v8f accG = splat8(biasG), accO = splat8(biasO);
#pragma unroll 4
    for (int kt = 0; kt < 16; ++kt) {
      const int ka = kt * 32 + aOff;
      v16bf a = ld_frag(aRow + ka, aRow + ka + 16);
      const int kb = kt * 32 + bOff;
      const unsigned short* p0 = &Wsh[(0 * 16 + colL) * WROW + kb];
      v16bf b0 = ld_frag(p0, p0 + 16);
      accF = __builtin_amdgcn_wmma_f32_16x16x32_bf16(false, a, false, b0,
                                                     (short)0, accF, false, false);
      const unsigned short* p1 = &Wsh[(1 * 16 + colL) * WROW + kb];
      v16bf b1 = ld_frag(p1, p1 + 16);
      accI = __builtin_amdgcn_wmma_f32_16x16x32_bf16(false, a, false, b1,
                                                     (short)0, accI, false, false);
      const unsigned short* p2 = &Wsh[(2 * 16 + colL) * WROW + kb];
      v16bf b2 = ld_frag(p2, p2 + 16);
      accG = __builtin_amdgcn_wmma_f32_16x16x32_bf16(false, a, false, b2,
                                                     (short)0, accG, false, false);
      const unsigned short* p3 = &Wsh[(3 * 16 + colL) * WROW + kb];
      v16bf b3 = ld_frag(p3, p3 + 16);
      accO = __builtin_amdgcn_wmma_f32_16x16x32_bf16(false, a, false, b3,
                                                     (short)0, accO, false, false);
    }

    // --- gate nonlinearities + state update (c stays in VGPRs) ---
    float* outT = out + (size_t)t * (BATCH * DH);
#pragma unroll
    for (int r2 = 0; r2 < 8; ++r2) {
      float fv = sigm(accF[r2]);
      float iv = sigm(accI[r2]);
      float gv = tanh_f(accG[r2]);
      float ov = sigm(accO[r2]);
      float cv = fv * cfrag[r2] + iv * gv;
      cfrag[r2] = cv;
      float hv = ov * tanh_f(cv);
      hlast[r2] = hv;
      const int row = rowBase + r2;
      outT[row * DH + col] = hv;            // fp32 sequence output
      hnxt[row * DH + col] = f2bf(hv);      // bf16 broadcast for next step
    }

    grid_barrier(&syncv[0], &syncv[1], t + 1);
  }

  // --- final (hx, cx) ---
  float* hxOut = out + (size_t)T_STEPS * BATCH * DH;
  float* cxOut = hxOut + BATCH * DH;
#pragma unroll
  for (int r2 = 0; r2 < 8; ++r2) {
    const int row = rowBase + r2;
    hxOut[row * DH + col] = hlast[r2];
    cxOut[row * DH + col] = cfrag[r2];
  }
}

// ---------------------------------------------------------------------------
extern "C" void kernel_launch(void* const* d_in, const int* in_sizes, int n_in,
                              void* d_out, int out_size, void* d_ws, size_t ws_size,
                              hipStream_t stream) {
  (void)in_sizes; (void)n_in; (void)out_size; (void)ws_size;
  const float* x   = (const float*)d_in[0];
  const float* Wf  = (const float*)d_in[1];
  const float* bfp = (const float*)d_in[2];
  const float* Wi  = (const float*)d_in[3];
  const float* bip = (const float*)d_in[4];
  const float* Wg  = (const float*)d_in[5];
  const float* bgp = (const float*)d_in[6];
  const float* Wo  = (const float*)d_in[7];
  const float* bop = (const float*)d_in[8];
  float* out = (float*)d_out;

  // Workspace layout: bf16 Wt [4][256][512] | bf16 h double buffer | sync ints
  unsigned short* wt = (unsigned short*)d_ws;                 // 1 MiB
  unsigned short* hb = wt + 4 * 256 * 512;                    // 128 KiB
  int* syncv = (int*)(hb + 2 * BATCH * DH);

  lstm_prep<<<512, 256, 0, stream>>>(Wf, Wi, Wg, Wo, wt, (unsigned int*)hb, syncv);
  lstm_main<<<NWG, THREADS, 0, stream>>>(x, bfp, bip, bgp, bop, wt, hb, syncv, out);
}
